// RNNModel_39092792328486
// MI455X (gfx1250) — compile-verified, matrix operand order checked by reading, not run on hardware
//
#include <hip/hip_runtime.h>
#include <hip/hip_bf16.h>

typedef float  v2f   __attribute__((ext_vector_type(2)));
typedef float  v8f   __attribute__((ext_vector_type(8)));
typedef __bf16 v8bf  __attribute__((ext_vector_type(8)));
typedef __bf16 v16bf __attribute__((ext_vector_type(16)));

#define VOCAB 10000
#define HID   256
#define BATCH 32
#define STEPS 256
#define NTILES (VOCAB / 16)      // 625 = 156*4 + 1

#define SW_STRIDE 264   // W_hh LDS row stride (floats): 2-row shift = 528%64 = 16 banks -> half-waves disjoint
#define SH_STRIDE 260   // h LDS row stride (floats): per-row shift = 4 banks -> 16 rows disjoint for b64 loads

// ---------------------------------------------------------------------------
// Kernel 0: transpose + convert  W_d [256][10000] f32  ->  W_dT [10000][256] bf16
// Makes B-matrix columns contiguous so WMMA bf16 B-fragments are 2x b128 loads.
// ---------------------------------------------------------------------------
__global__ void wd_transpose_kernel(const float* __restrict__ Wd,
                                    __bf16* __restrict__ WdT) {
  __shared__ float tile[32][33];
  const int tx = threadIdx.x, ty = threadIdx.y;       // 32 x 8
  const int nbase = blockIdx.x * 32;
  const int kbase = blockIdx.y * 32;
#pragma unroll
  for (int i = 0; i < 4; ++i) {
    int k = kbase + ty + i * 8;
    int n = nbase + tx;
    float v = (n < VOCAB) ? Wd[k * VOCAB + n] : 0.f;
    tile[ty + i * 8][tx] = v;
  }
  __syncthreads();
#pragma unroll
  for (int i = 0; i < 4; ++i) {
    int n = nbase + ty + i * 8;
    int k = kbase + tx;
    if (n < VOCAB) WdT[n * HID + k] = (__bf16)tile[tx][ty + i * 8];
  }
}

// ---------------------------------------------------------------------------
// Kernel 1: fused gather + recurrent scan (f32 WMMA 16x16x4), W_hh LDS-resident.
// 2 blocks x 512 threads (16 waves). Block owns 16 batch rows; wave owns one
// 16-wide N-tile of h. Emits Y as bf16 for the output GEMM; final h as f32.
// Latency-bound: 256 dependent steps; K split into two accumulator chains.
// ---------------------------------------------------------------------------
__global__ void __launch_bounds__(512)
rnn_scan_kernel(const int*   __restrict__ tokens,
                const float* __restrict__ Wxh,
                const float* __restrict__ Whh,
                const float* __restrict__ bh,
                const float* __restrict__ h0,
                __bf16*      __restrict__ Yb,
                float*       __restrict__ stateOut) {
  extern __shared__ float smem[];
  float* sW   = smem;                       // [256][SW_STRIDE]
  float* sh   = sW + HID * SW_STRIDE;       // [16][SH_STRIDE]
  float* sbh  = sh + 16 * SH_STRIDE;        // [256]
  int*   sTok = (int*)(sbh + HID);          // [16]

  const int tid     = threadIdx.x;
  const int lane    = tid & 31;
  const int wave    = tid >> 5;             // 0..15 -> n-tile
  const int b0      = blockIdx.x * 16;      // batch-row base
  const int halfsel = lane >> 4;            // 0|1 (half-wave)
  const int lcol    = lane & 15;
  const int col     = wave * 16 + lcol;     // hidden column 0..255
  const int kk0     = 2 * halfsel;          // f32 A/B frag K sub-offset

  // ---- stage W_hh, h0, b_h into LDS (amortized over 256 steps) ----
  for (int i = tid; i < HID * HID; i += 512) {
    int k = i >> 8, n = i & 255;
    sW[k * SW_STRIDE + n] = Whh[i];
  }
  for (int i = tid; i < 16 * HID; i += 512) {
    int m = i >> 8, c = i & 255;
    sh[m * SH_STRIDE + c] = h0[(b0 + m) * HID + c];
  }
  for (int i = tid; i < HID; i += 512) sbh[i] = bh[i];
  __syncthreads();

  for (int t = 0; t < STEPS; ++t) {
    if (tid < 16) sTok[tid] = tokens[(b0 + tid) * STEPS + t];
    __syncthreads();  // sTok ready; previous step's sh writes visible

    // Prefetch x_t = W_xh[token] for this wave's columns; consumed after the
    // WMMA loop so the gather latency hides under the recurrent GEMM.
    float xw[8];
#pragma unroll
    for (int v = 0; v < 8; ++v) {
      int m = v + 8 * halfsel;                       // C-layout row for vgpr v
      xw[v] = Wxh[(long)sTok[m] * HID + col];
    }

    // h_{t-1} @ W_hh  with K split into two chains (shorter dependent chain)
    v8f acc0 = {}, acc1 = {};
#pragma unroll 8
    for (int k0 = 0; k0 < 128; k0 += 4) {
      v2f a = *(const v2f*)(sh + lcol * SH_STRIDE + k0 + kk0);
      v2f b;
      b.x = sW[(k0 + kk0)     * SW_STRIDE + col];
      b.y = sW[(k0 + kk0 + 1) * SW_STRIDE + col];
      acc0 = __builtin_amdgcn_wmma_f32_16x16x4_f32(false, a, false, b,
                                                   (short)0, acc0, false, false);
    }
#pragma unroll 8
    for (int k0 = 128; k0 < 256; k0 += 4) {
      v2f a = *(const v2f*)(sh + lcol * SH_STRIDE + k0 + kk0);
      v2f b;
      b.x = sW[(k0 + kk0)     * SW_STRIDE + col];
      b.y = sW[(k0 + kk0 + 1) * SW_STRIDE + col];
      acc1 = __builtin_amdgcn_wmma_f32_16x16x4_f32(false, a, false, b,
                                                   (short)0, acc1, false, false);
    }

    const float bias = sbh[col];
    float hn[8];
#pragma unroll
    for (int v = 0; v < 8; ++v)
      hn[v] = tanhf(acc0[v] + acc1[v] + xw[v] + bias);

    __syncthreads();  // all reads of sh/sTok for step t complete

#pragma unroll
    for (int v = 0; v < 8; ++v) {
      int m = v + 8 * halfsel;
      sh[m * SH_STRIDE + col] = hn[v];                       // h_t for next step
      long row = (long)t * BATCH + (b0 + m);                 // [T,B] flatten order
      Yb[row * HID + col] = (__bf16)hn[v];
      if (t == STEPS - 1) stateOut[(b0 + m) * HID + col] = hn[v];
    }
  }
}

// ---------------------------------------------------------------------------
// Kernel 2: output projection  out[8192,10000] = Y @ W_d + b_d
// bf16 WMMA 16x16x32, f32 accumulate. Wave computes a 16x64 tile (A reuse x4).
// Store-bandwidth bound: 327.7 MB f32 NT writes -> ~14 us floor at 23.3 TB/s.
// Main kernel is guard-free (n-tiles 0..623); a tail kernel does n-tile 624.
// ---------------------------------------------------------------------------
__device__ __forceinline__ v16bf load_frag_bf16(const __bf16* __restrict__ p) {
  v8bf lo = *(const v8bf*)(p);        // halves 0-7  : K = 8*(lane/16) + 0..7
  v8bf hi = *(const v8bf*)(p + 16);   // halves 8-15 : K = 8*(lane/16) + 16..23
  return __builtin_shufflevector(lo, hi, 0, 1, 2, 3, 4, 5, 6, 7,
                                         8, 9, 10, 11, 12, 13, 14, 15);
}

__global__ void __launch_bounds__(256)
out_gemm_kernel(const __bf16* __restrict__ Yb,
                const __bf16* __restrict__ WdT,
                const float*  __restrict__ bd,
                float*        __restrict__ out) {
  const int tid     = threadIdx.x;
  const int lane    = tid & 31;
  const int wave    = tid >> 5;                 // 0..7
  const int lcol    = lane & 15;
  const int halfsel = lane >> 4;
  const int mtile   = blockIdx.y * 8 + wave;    // 0..511
  const int ntbase  = blockIdx.x * 4;           // n-tile base; always full (0..623)
  const int koff    = 8 * halfsel;              // 16-bit A/B frag K sub-offset

  const long abase = (long)(mtile * 16 + lcol) * HID;
  const long bbase = (long)(ntbase * 16 + lcol) * HID;

  v8f acc[4] = {v8f{}, v8f{}, v8f{}, v8f{}};
#pragma unroll
  for (int k0 = 0; k0 < HID; k0 += 32) {
    v16bf a = load_frag_bf16(Yb + abase + k0 + koff);
#pragma unroll
    for (int j = 0; j < 4; ++j) {
      v16bf b = load_frag_bf16(WdT + bbase + (long)j * 16 * HID + k0 + koff);
      acc[j] = __builtin_amdgcn_wmma_f32_16x16x32_bf16(false, a, false, b,
                                                       (short)0, acc[j],
                                                       false, false);
    }
  }

#pragma unroll
  for (int j = 0; j < 4; ++j) {
    int n = (ntbase + j) * 16 + lcol;
    float bias = bd[n];
#pragma unroll
    for (int v = 0; v < 8; ++v) {
      int row = mtile * 16 + v + 8 * halfsel;   // C-layout: M = v + 8*(lane/16)
      __builtin_nontemporal_store(acc[j][v] + bias, out + (long)row * VOCAB + n);
    }
  }
}

// Tail: n-tile 624 (columns 9984..9999) for all 512 m-tiles.
__global__ void __launch_bounds__(256)
out_gemm_tail_kernel(const __bf16* __restrict__ Yb,
                     const __bf16* __restrict__ WdT,
                     const float*  __restrict__ bd,
                     float*        __restrict__ out) {
  const int tid     = threadIdx.x;
  const int lane    = tid & 31;
  const int wave    = tid >> 5;
  const int lcol    = lane & 15;
  const int halfsel = lane >> 4;
  const int mtile   = blockIdx.x * 8 + wave;    // 0..511
  const int nt      = NTILES - 1;               // 624
  const int koff    = 8 * halfsel;

  const long abase = (long)(mtile * 16 + lcol) * HID;
  const long bbase = (long)(nt * 16 + lcol) * HID;

  v8f acc = {};
#pragma unroll
  for (int k0 = 0; k0 < HID; k0 += 32) {
    v16bf a = load_frag_bf16(Yb + abase + k0 + koff);
    v16bf b = load_frag_bf16(WdT + bbase + k0 + koff);
    acc = __builtin_amdgcn_wmma_f32_16x16x32_bf16(false, a, false, b,
                                                  (short)0, acc, false, false);
  }

  const int n = nt * 16 + lcol;
  const float bias = bd[n];
#pragma unroll
  for (int v = 0; v < 8; ++v) {
    int row = mtile * 16 + v + 8 * halfsel;
    __builtin_nontemporal_store(acc[v] + bias, out + (long)row * VOCAB + n);
  }
}

// ---------------------------------------------------------------------------
extern "C" void kernel_launch(void* const* d_in, const int* in_sizes, int n_in,
                              void* d_out, int out_size, void* d_ws, size_t ws_size,
                              hipStream_t stream) {
  const int*   tokens = (const int*)  d_in[0];
  const float* Wxh    = (const float*)d_in[1];
  const float* Whh    = (const float*)d_in[2];
  const float* bh     = (const float*)d_in[3];
  const float* Wd     = (const float*)d_in[4];
  const float* bd     = (const float*)d_in[5];
  const float* h0     = (const float*)d_in[6];

  float* out      = (float*)d_out;
  float* stateOut = out + (long)STEPS * BATCH * VOCAB;   // outputs concat: [logits | state]

  __bf16* Yb  = (__bf16*)d_ws;                                            // 4,194,304 B
  __bf16* WdT = (__bf16*)((char*)d_ws + (size_t)STEPS * BATCH * HID * 2); // 5,120,000 B

  // 0) W_d -> bf16 transpose
  wd_transpose_kernel<<<dim3((VOCAB + 31) / 32, HID / 32), dim3(32, 8), 0, stream>>>(Wd, WdT);

  // 1) fused gather + scan; W_hh LDS-resident (needs 288,064 B dynamic LDS/WGP)
  size_t lds = (size_t)(HID * SW_STRIDE + 16 * SH_STRIDE + HID) * sizeof(float)
             + 16 * sizeof(int);
  rnn_scan_kernel<<<BATCH / 16, 512, lds, stream>>>(tokens, Wxh, Whh, bh, h0, Yb, stateOut);

  // 2) output projection (bf16 WMMA, f32 accumulate + bias, NT stores)
  out_gemm_kernel<<<dim3(NTILES / 4, (STEPS * BATCH / 16) / 8), 256, 0, stream>>>(
      Yb, WdT, bd, out);
  out_gemm_tail_kernel<<<(STEPS * BATCH / 16) / 8, 256, 0, stream>>>(Yb, WdT, bd, out);
}
